// AblationModel_16020228014624
// MI455X (gfx1250) — compile-verified, matrix operand order checked by reading, not run on hardware
//
#include <hip/hip_runtime.h>
#include <hip/hip_bf16.h>

// ---------------- problem constants (match reference) ----------------
#define T_     256
#define TM_    255
#define Z_     512
#define A_     5
#define D_     8192
#define TOPK_  50
#define DELTA_ 0.001f
#define WPITCH 8448           // D_ + TM_ = 8447, padded
#define NW_    (D_ + TM_)     // 8447 candidates per row
#define NCOL_  64             // B columns staged in LDS per block (64 KB f16)

typedef __attribute__((ext_vector_type(16))) _Float16 v16h;
typedef __attribute__((ext_vector_type(8)))  _Float16 v8h;
typedef __attribute__((ext_vector_type(8)))  float    v8f;
typedef __attribute__((ext_vector_type(4)))  unsigned int v4u;
typedef __attribute__((ext_vector_type(4)))  int      v4i;
typedef __attribute__((ext_vector_type(8)))  int      v8i;

#ifndef __has_builtin
#define __has_builtin(x) 0
#endif
#if __has_builtin(__builtin_amdgcn_tensor_load_to_lds)
#define USE_TDM 1
#else
#define USE_TDM 0
#endif

// ------------------------------------------------------------------
// Build per-action stable ordering of timesteps. rank within group ==
// cnt[t] (number of prior same-action steps) by construction.
// ------------------------------------------------------------------
__global__ void build_index(const int* __restrict__ actions,
                            int* __restrict__ ord,
                            int* __restrict__ gstart,
                            int* __restrict__ gcnt) {
    if (threadIdx.x == 0 && blockIdx.x == 0) {
        int pos = 0;
        for (int al = 0; al < A_; ++al) {
            gstart[al] = pos;
            for (int t = 0; t < TM_; ++t)
                if (actions[t + 1] == al) ord[pos++] = t;
            gcnt[al] = pos - gstart[al];
        }
    }
}

// ------------------------------------------------------------------
// Row-wise sum of squares: one wave (32 lanes) per row of length Z_.
// ------------------------------------------------------------------
__global__ void rowsumsq(const float* __restrict__ src, int nrows,
                         float* __restrict__ out) {
    int gwave = (blockIdx.x * blockDim.x + threadIdx.x) >> 5;
    int lane  = threadIdx.x & 31;
    if (gwave >= nrows) return;
    const float* row = src + (size_t)gwave * Z_;
    float s = 0.f;
    for (int z = lane; z < Z_; z += 32) s = fmaf(row[z], row[z], s);
    #pragma unroll
    for (int off = 16; off >= 1; off >>= 1) s += __shfl_xor(s, off, 32);
    if (lane == 0) out[gwave] = s;
}

// ------------------------------------------------------------------
// Bulk f32 -> f16 conversion, 8 elements per thread (b128 in, b128 out).
// ------------------------------------------------------------------
__global__ void cvt_f32_to_f16(const float* __restrict__ src,
                               _Float16* __restrict__ dst, int n8) {
    int i = blockIdx.x * blockDim.x + threadIdx.x;
    if (i >= n8) return;
    const float4* p = (const float4*)src + (size_t)i * 2;
    float4 a = p[0], b = p[1];
    v8h o;
    o[0] = (_Float16)a.x; o[1] = (_Float16)a.y;
    o[2] = (_Float16)a.z; o[3] = (_Float16)a.w;
    o[4] = (_Float16)b.x; o[5] = (_Float16)b.y;
    o[6] = (_Float16)b.z; o[7] = (_Float16)b.w;
    ((v8h*)dst)[i] = o;
}

// ------------------------------------------------------------------
// 16-bit WMMA fragment from 8+8 consecutive halves (ISA 7.12.2 layout):
//   lanes 0-15 : elems 0..7 -> K=zoff+0..7,  8..15 -> K=zoff+16..23
//   lanes 16-31: elems 0..7 -> K=zoff+8..15, 8..15 -> K=zoff+24..31
// Works for global (qf16) and LDS (Bs) sources; 16-byte aligned.
// ------------------------------------------------------------------
__device__ __forceinline__ v16h frag_halves(const _Float16* p, int zoff,
                                            int khalf) {
    v8h lo = *(const v8h*)(p + zoff + khalf);
    v8h hi = *(const v8h*)(p + zoff + khalf + 16);
    v16h f;
    #pragma unroll
    for (int i = 0; i < 8; ++i) { f[i] = lo[i]; f[i + 8] = hi[i]; }
    return f;
}

// ------------------------------------------------------------------
// Grouped GEMM + w_init epilogue, LDS-staged B.
// blockIdx.x = action alpha; blockIdx.y = 64-column slab of D.
// The f16 B slab (64 x 512, 64 KB, contiguous in kf16) is staged into
// LDS once per block via the Tensor Data Mover, then reused by every
// 16-row tile of the action group. Wave w: columns (w&3)*16..+15,
// row-tiles m = (w>>2), (w>>2)+2, ...
// ------------------------------------------------------------------
__global__ void __launch_bounds__(256)
gemm_winit(const _Float16* __restrict__ qf16,
           const _Float16* __restrict__ kf16,
           const int*   __restrict__ ord,
           const int*   __restrict__ gstart,
           const int*   __restrict__ gcnt_,
           const float* __restrict__ qsq,
           const float* __restrict__ ksq,
           float* __restrict__ w_all) {
    __shared__ __align__(128) _Float16 Bs[NCOL_ * Z_];   // 64 KB

    const int alpha = blockIdx.x;
    const int slab  = blockIdx.y;                         // 0..D_/NCOL_-1
    const int gcnt  = gcnt_[alpha];
    if (gcnt == 0) return;                                // uniform
    const int gbase = gstart[alpha];

    const _Float16* src = kf16 + ((size_t)alpha * D_ + (size_t)slab * NCOL_) * Z_;

#if USE_TDM
    // ---- TDM: one wave issues a 2-D tile load (512 x 64 halves) ----
    if (threadIdx.x < 32) {
        unsigned long long ga = (unsigned long long)(uintptr_t)src;
        unsigned int ldsoff   = (unsigned int)(size_t)(&Bs[0]);
        v4u g0;
        g0[0] = 1u;                                   // count=1, user D#
        g0[1] = ldsoff;                               // lds_addr
        g0[2] = (unsigned int)ga;                     // global_addr[31:0]
        g0[3] = ((unsigned int)(ga >> 32) & 0x01FFFFFFu) | (2u << 30); // [56:32] | type=2
        v8i g1;
        g1[0] = (int)(1u << 16);          // wgmask=0, data_size=1 (2 bytes)
        g1[1] = (int)(512u << 16);        // tensor_dim0 = 512 (low16 in [31:16])
        g1[2] = (int)(40960u << 16);      // dim0 hi=0 | tensor_dim1 = A_*D_ rows
        g1[3] = (int)(512u << 16);        // dim1 hi=0 | tile_dim0 = 512
        g1[4] = NCOL_;                    // tile_dim1 = 64, tile_dim2 = 0
        g1[5] = 512;                      // tensor_dim0_stride = 512 (lo32)
        g1[6] = 0;                        // stride0 hi | stride1 lo
        g1[7] = 0;
        v4i g2 = {};                      // 2-D tensor: groups 2/3 unused
        v4i g3 = {};
#if __clang_major__ >= 23
        v8i g4 = {};
        __builtin_amdgcn_tensor_load_to_lds(g0, g1, g2, g3, g4, 0);
#else
        __builtin_amdgcn_tensor_load_to_lds(g0, g1, g2, g3, 0);
#endif
        __builtin_amdgcn_s_wait_tensorcnt(0);
    }
    __syncthreads();
#else
    for (int i = threadIdx.x; i < (NCOL_ * Z_) / 8; i += 256)
        ((v8h*)Bs)[i] = ((const v8h*)src)[i];
    __syncthreads();
#endif

    const int wave    = threadIdx.x >> 5;
    const int lane    = threadIdx.x & 31;
    const int mlane   = lane & 15;
    const int khalf   = (lane >> 4) << 3;                 // 0 or 8
    const int colbase = (wave & 3) * 16;
    const int dlane   = slab * NCOL_ + colbase + mlane;   // this lane's N column
    const _Float16* bcol = &Bs[(size_t)(colbase + mlane) * Z_];
    const float kk = ksq[alpha * D_ + dlane];
    const int mrow0 = (lane >> 4) << 3;

    for (int m = (wave >> 2); m * 16 < gcnt; m += 2) {
        int grow = m * 16 + mlane;
        int t_a  = ord[gbase + ((grow < gcnt) ? grow : 0)];
        const _Float16* qrow = qf16 + (size_t)t_a * Z_;

        v8f acc = {};
        #pragma unroll 4
        for (int zoff = 0; zoff < Z_; zoff += 32) {
            v16h a = frag_halves(qrow, zoff, khalf);
            v16h b = frag_halves(bcol, zoff, khalf);
            acc = __builtin_amdgcn_wmma_f32_16x16x32_f16(
                false, a, false, b, (short)0, acc, false, false);
        }

        // C layout: lane 0-15 -> M=v, lane 16-31 -> M=v+8 ; N=mlane
        #pragma unroll
        for (int v = 0; v < 8; ++v) {
            int g = m * 16 + mrow0 + v;          // rank within group == cnt[t]
            if (g < gcnt) {
                int tt   = ord[gbase + g];
                float d2 = qsq[tt] - 2.0f * acc[v] + kk;
                float w  = (dlane >= g) ? 1.0f / (d2 + DELTA_) : 0.0f;
                w_all[(size_t)tt * WPITCH + dlane] = w;
            }
        }
    }
}

// ------------------------------------------------------------------
// Pairwise same-action weights: w_all[t][D+j].
// ------------------------------------------------------------------
__global__ void wwrit(const float* __restrict__ emb,
                      const int*   __restrict__ actions,
                      const float* __restrict__ qsq,
                      float* __restrict__ w_all) {
    int t  = blockIdx.x;                     // 0..254
    int at = actions[t + 1];
    const float* qa = emb + (size_t)t * Z_;
    for (int j = threadIdx.x; j < TM_; j += blockDim.x) {
        float w = 0.f;
        if (j < t && actions[j + 1] == at) {
            const float* qb = emb + (size_t)j * Z_;
            float dot = 0.f;
            for (int z = 0; z < Z_; ++z) dot = fmaf(qa[z], qb[z], dot);
            float d2 = qsq[t] - 2.f * dot + qsq[j];
            w = 1.0f / (d2 + DELTA_);
        }
        w_all[(size_t)t * WPITCH + D_ + j] = w;
    }
}

// ------------------------------------------------------------------
// Top-50 selection + fused weighted gather of value vectors.
// One block (256 threads) per row t; 50 rounds of block argmax.
// Winner masked with -1 (valid weights >= 0); w_all fully rebuilt
// every call so this is replay-safe.
// ------------------------------------------------------------------
__global__ void topk_gather(float* __restrict__ w_all,
                            const float* __restrict__ emb,
                            const float* __restrict__ mem_vals,
                            const int*   __restrict__ actions,
                            float* __restrict__ pred_emb) {
    __shared__ float smax[256];
    __shared__ int   simax[256];
    __shared__ float sel_w;
    __shared__ int   sel_i;
    __shared__ float swsum;

    int t   = blockIdx.x;
    int tid = threadIdx.x;
    int at  = actions[t + 1];
    float* row = w_all + (size_t)t * WPITCH;
    float e0 = 0.f, e1 = 0.f;                // z = tid, tid+256
    if (tid == 0) swsum = 0.f;
    __syncthreads();

    for (int k = 0; k < TOPK_; ++k) {
        float mv = -1e30f; int mi = 0;
        for (int i = tid; i < NW_; i += 256) {
            float v = row[i];
            if (v > mv) { mv = v; mi = i; }
        }
        smax[tid] = mv; simax[tid] = mi;
        __syncthreads();
        #pragma unroll
        for (int s = 128; s > 0; s >>= 1) {
            if (tid < s) {
                bool take = (smax[tid + s] > smax[tid]) ||
                            (smax[tid + s] == smax[tid] &&
                             simax[tid + s] < simax[tid]);
                if (take) { smax[tid] = smax[tid + s]; simax[tid] = simax[tid + s]; }
            }
            __syncthreads();
        }
        if (tid == 0) {
            sel_w = smax[0];
            sel_i = simax[0];
            row[simax[0]] = -1.0f;
            swsum += smax[0];
        }
        __syncthreads();
        float w  = sel_w;
        int  idx = sel_i;
        const float* vsrc = (idx < D_)
            ? (mem_vals + ((size_t)at * D_ + idx) * Z_)
            : (emb + (size_t)(idx - D_ + 1) * Z_);
        e0 = fmaf(w, vsrc[tid], e0);
        e1 = fmaf(w, vsrc[tid + 256], e1);
        __syncthreads();
    }
    float inv = 1.0f / swsum;
    pred_emb[(size_t)t * Z_ + tid]       = e0 * inv;
    pred_emb[(size_t)t * Z_ + tid + 256] = e1 * inv;
}

// ------------------------------------------------------------------
// MLP: 512 -> 256 (elu) -> 128 (elu) -> 1. One wave per row, 8 waves
// per block; intermediate activations staged in LDS.
// ------------------------------------------------------------------
__global__ void mlp(const float* __restrict__ X, int n,
                    const float* __restrict__ W1, const float* __restrict__ b1,
                    const float* __restrict__ W2, const float* __restrict__ b2,
                    const float* __restrict__ W3, const float* __restrict__ b3,
                    float* __restrict__ out) {
    __shared__ float h1s[8][256];
    __shared__ float h2s[8][128];
    int wave = threadIdx.x >> 5;
    int lane = threadIdx.x & 31;
    int row  = blockIdx.x * 8 + wave;
    bool active = row < n;
    const float* x = X + (size_t)(active ? row : 0) * Z_;

    float h[8];
    #pragma unroll
    for (int u = 0; u < 8; ++u) h[u] = b1[lane * 8 + u];
    for (int z = 0; z < Z_; ++z) {
        float xv = x[z];
        const float* wr = W1 + (size_t)z * 256 + lane * 8;
        #pragma unroll
        for (int u = 0; u < 8; ++u) h[u] = fmaf(xv, wr[u], h[u]);
    }
    #pragma unroll
    for (int u = 0; u < 8; ++u) {
        float v = h[u];
        h1s[wave][lane * 8 + u] = (v > 0.f) ? v : (__expf(v) - 1.0f);
    }
    __syncthreads();

    float g[4];
    #pragma unroll
    for (int u = 0; u < 4; ++u) g[u] = b2[lane * 4 + u];
    for (int z = 0; z < 256; ++z) {
        float hv = h1s[wave][z];
        const float* wr = W2 + (size_t)z * 128 + lane * 4;
        #pragma unroll
        for (int u = 0; u < 4; ++u) g[u] = fmaf(hv, wr[u], g[u]);
    }
    #pragma unroll
    for (int u = 0; u < 4; ++u) {
        float v = g[u];
        h2s[wave][lane * 4 + u] = (v > 0.f) ? v : (__expf(v) - 1.0f);
    }
    __syncthreads();

    float s = 0.f;
    #pragma unroll
    for (int u = 0; u < 4; ++u)
        s = fmaf(h2s[wave][lane * 4 + u], W3[lane * 4 + u], s);
    #pragma unroll
    for (int off = 16; off >= 1; off >>= 1) s += __shfl_xor(s, off, 32);
    if (lane == 0 && active) out[row] = s + b3[0];
}

// ------------------------------------------------------------------
// Final scalar loss.
// ------------------------------------------------------------------
__global__ void loss_kernel(const float* __restrict__ pv,
                            const float* __restrict__ pt,
                            const float* __restrict__ values,
                            float* __restrict__ out) {
    __shared__ float red[256];
    int tid = threadIdx.x;
    float s;
    {
        float d = pv[tid] - values[tid];
        s = d * d * (1.0f / 256.0f);
    }
    if (tid < TM_) {
        float d = pt[tid] - values[tid + 1];
        s = fmaf(d * d, 1.0f / 255.0f, s);
    }
    red[tid] = s;
    __syncthreads();
    #pragma unroll
    for (int st = 128; st > 0; st >>= 1) {
        if (tid < st) red[tid] += red[tid + st];
        __syncthreads();
    }
    if (tid == 0) out[0] = red[0];
}

// ------------------------------------------------------------------
extern "C" void kernel_launch(void* const* d_in, const int* in_sizes, int n_in,
                              void* d_out, int out_size, void* d_ws, size_t ws_size,
                              hipStream_t stream) {
    const float* emb      = (const float*)d_in[0];   // (256,512)
    const float* values   = (const float*)d_in[1];   // (256,)
    const float* mem_keys = (const float*)d_in[2];   // (5,8192,512)
    const float* mem_vals = (const float*)d_in[3];   // (5,8192,512)
    const float* W1       = (const float*)d_in[4];   // (512,256)
    const float* b1       = (const float*)d_in[5];
    const float* W2       = (const float*)d_in[6];   // (256,128)
    const float* b2       = (const float*)d_in[7];
    const float* W3       = (const float*)d_in[8];   // (128,1)
    const float* b3       = (const float*)d_in[9];
    const int*   actions  = (const int*)d_in[10];    // (256,)
    float* out = (float*)d_out;
    (void)in_sizes; (void)n_in; (void)out_size; (void)ws_size;

    // workspace layout
    float* wsf = (float*)d_ws;
    int*   wsi = (int*)d_ws;
    int*   ord         = wsi;                 // 256 ints
    int*   gstart      = wsi + 256;           // 8
    int*   gcnt        = wsi + 264;           // 8
    float* qsq         = wsf + 512;           // 256
    float* pred_values = wsf + 768;           // 256
    float* pred_t      = wsf + 1024;          // 256
    float* ksq         = wsf + 1280;          // 40960    -> ends 42240
    float* pred_emb    = wsf + 43008;         // 255*512  -> ends 173568
    float* w_all       = wsf + 174080;        // 255*8448 -> ends 2328320
    _Float16* kf16     = (_Float16*)(wsf + 2359296);   // 5*8192*512 halves (40 MB)
    _Float16* qf16     = (_Float16*)(wsf + 12845056);  // 255*512 halves

    build_index<<<1, 32, 0, stream>>>(actions, ord, gstart, gcnt);

    rowsumsq<<<(TM_ + 7) / 8, 256, 0, stream>>>(emb, TM_, qsq);
    rowsumsq<<<(A_ * D_ + 7) / 8, 256, 0, stream>>>(mem_keys, A_ * D_, ksq);

    // one-shot f32 -> f16 conversions (keys: 80 MB read, 40 MB write)
    cvt_f32_to_f16<<<(A_ * D_ * Z_ / 8 + 255) / 256, 256, 0, stream>>>(
        mem_keys, kf16, A_ * D_ * Z_ / 8);
    cvt_f32_to_f16<<<(TM_ * Z_ / 8 + 255) / 256, 256, 0, stream>>>(
        emb, qf16, TM_ * Z_ / 8);

    // grouped WMMA GEMM, TDM-staged B slabs
    {
        dim3 grid(A_, D_ / NCOL_, 1);
        gemm_winit<<<grid, 256, 0, stream>>>(qf16, kf16, ord, gstart, gcnt,
                                             qsq, ksq, w_all);
    }

    wwrit<<<TM_, 256, 0, stream>>>(emb, actions, qsq, w_all);

    topk_gather<<<TM_, 256, 0, stream>>>(w_all, emb, mem_vals, actions, pred_emb);

    mlp<<<(T_ + 7) / 8, 256, 0, stream>>>(emb, T_, W1, b1, W2, b2, W3, b3,
                                          pred_values);
    mlp<<<(TM_ + 7) / 8, 256, 0, stream>>>(pred_emb, TM_, W1, b1, W2, b2, W3, b3,
                                           pred_t);

    loss_kernel<<<1, 256, 0, stream>>>(pred_values, pred_t, values, out);
}
